// REEHeteroGAT_10840497455106
// MI455X (gfx1250) — matrix operand order repaired and checked by reading, not compiled, not run on hardware
//
#include <hip/hip_runtime.h>
#include <hip/hip_bf16.h>
#include <stdint.h>

typedef __attribute__((ext_vector_type(16))) _Float16 v16h;
typedef __attribute__((ext_vector_type(8)))  float    v8f;

// ---------- helpers ----------
__device__ __forceinline__ unsigned fmax_key(float f) {
  unsigned u = __float_as_uint(f);
  return (u & 0x80000000u) ? ~u : (u | 0x80000000u);
}
__device__ __forceinline__ float fmax_unkey(unsigned k) {
  unsigned u = (k & 0x80000000u) ? (k ^ 0x80000000u) : ~k;
  return __uint_as_float(u);
}

// ---------- elementwise / small kernels ----------
__global__ void k_cvt_h(const float* __restrict__ in, _Float16* __restrict__ out, int n) {
  int i = blockIdx.x * blockDim.x + threadIdx.x;
  if (i < n) out[i] = (_Float16)in[i];
}

__global__ void k_gather_emb(const int* __restrict__ idx, const float* __restrict__ emb,
                             float* __restrict__ xf, _Float16* __restrict__ xh, int n) {
  int i = blockIdx.x * blockDim.x + threadIdx.x;
  if (i >= n * 128) return;
  int node = i >> 7, o = i & 127;
  float v = emb[(size_t)idx[node] * 128 + o];
  xf[i] = v; xh[i] = (_Float16)v;
}

// relu(x @ W1 + b1) for tiny K (f in {1,3,5}) -> f16 hidden
__global__ void k_enc1(const float* __restrict__ x, const float* __restrict__ W1,
                       const float* __restrict__ b1, _Float16* __restrict__ h, int n, int f) {
  int i = blockIdx.x * blockDim.x + threadIdx.x;
  if (i >= n * 128) return;
  int node = i >> 7, o = i & 127;
  float acc = b1[o];
  for (int k = 0; k < f; ++k) acc += x[(size_t)node * f + k] * W1[k * 128 + o];
  h[i] = (_Float16)(acc > 0.f ? acc : 0.f);
}

// ---------- WMMA GEMM: C[M,N] = A[M,128] @ W[128,N] (+bias, +relu), K=128 ----------
// MODE 0: plain; MODE 1: +bias; MODE 2: +bias+relu. Stores Cf32/Ch16 if non-null.
template <int MODE>
__global__ __launch_bounds__(256) void k_gemm(const _Float16* __restrict__ A,
                                              const _Float16* __restrict__ W,
                                              const float* __restrict__ bias,
                                              float* __restrict__ C, _Float16* __restrict__ Ch,
                                              int M, int N) {
  static __shared__ __attribute__((aligned(32))) _Float16 lds[16384]; // up to 4 ksteps * 8 ntiles * 512 halves
  const int NT = N >> 4;
  const int tid = threadIdx.x;
  // Stage W into LDS pre-swizzled into the 32x16 B-fragment layout:
  // frag[kstep][ntile][lane][t] = W[kstep*32 + (lane>=16 ? 16:0) + t][ntile*16 + (lane&15)]
  const int total = 4 * NT * 32 * 16;
  for (int idx = tid; idx < total; idx += 256) {
    int t = idx & 15;
    int lane = (idx >> 4) & 31;
    int rem = idx >> 9;              // kstep*NT + ntile
    int ntile = rem % NT;
    int kstep = rem / NT;
    int k = kstep * 32 + ((lane >> 4) << 4) + t;
    int n = ntile * 16 + (lane & 15);
    lds[idx] = W[(size_t)k * N + n];
  }
  __syncthreads();
  const int wave = tid >> 5, lane = tid & 31;
  const int m0 = blockIdx.x * 128 + wave * 16;
  if (m0 >= M) return;
  const int mrow = m0 + (lane & 15);
  const _Float16* ap = A + (size_t)(mrow < M ? mrow : 0) * 128;
  // A fragments: 16x32 f16 layout; lane<16 holds K = ks*32 + {0..15}; lane>=16 holds +16.
  const int koff = (lane >= 16) ? 16 : 0;
  v16h afrag[4];
#pragma unroll
  for (int ks = 0; ks < 4; ++ks) {
    int K0 = ks * 32 + koff;
    v16h a;
#pragma unroll
    for (int t = 0; t < 8; ++t) { a[t] = ap[K0 + t]; a[t + 8] = ap[K0 + 8 + t]; }
    afrag[ks] = a;
  }
  const int rbase = m0 + ((lane >> 4) << 3);
  for (int nt = 0; nt < NT; ++nt) {
    v8f c = {};
#pragma unroll
    for (int ks = 0; ks < 4; ++ks) {
      const v16h* bp = (const v16h*)&lds[((ks * NT + nt) * 32 + lane) * 16];
      c = __builtin_amdgcn_wmma_f32_16x16x32_f16(false, afrag[ks], false, *bp,
                                                 (short)0, c, false, false);
    }
    int ncol = nt * 16 + (lane & 15);
    float bv = (MODE >= 1 && bias) ? bias[ncol] : 0.f;
#pragma unroll
    for (int i = 0; i < 8; ++i) {
      int r = rbase + i;
      if (r < M) {
        float v = c[i] + bv;
        if (MODE == 2) v = v > 0.f ? v : 0.f;
        if (C)  C[(size_t)r * N + ncol] = v;
        if (Ch) Ch[(size_t)r * N + ncol] = (_Float16)v;
      }
    }
  }
}

// ---------- GAT attention kernels ----------
// a_s[n,h] = sum_c hs[n, h*32+c] * att_s[h,c]   (att flat over k = h*32+c)
__global__ void k_as(const float* __restrict__ hs, const float* __restrict__ att,
                     float* __restrict__ as_, int n) {
  int wave = threadIdx.x >> 5, lane = threadIdx.x & 31;
  int node = blockIdx.x * 8 + wave;
  if (node >= n) return;
  int k0 = lane * 4;
  float p = 0.f;
#pragma unroll
  for (int j = 0; j < 4; ++j) p += hs[(size_t)node * 128 + k0 + j] * att[k0 + j];
  p += __shfl_xor(p, 1); p += __shfl_xor(p, 2); p += __shfl_xor(p, 4);
  if ((lane & 7) == 0) as_[node * 4 + (lane >> 3)] = p;
}

// wdv[k,h] = sum_c Wd[k, h*32+c] * att_d[h,c]   (rank-4 fold of the Wd GEMM)
__global__ void k_wdv(const float* __restrict__ Wd, const float* __restrict__ attd,
                      float* __restrict__ wdv) {
  int i = threadIdx.x;
  if (i >= 512) return;
  int k = i >> 2, h = i & 3;
  float s = 0.f;
  for (int c = 0; c < 32; ++c) s += Wd[k * 128 + h * 32 + c] * attd[h * 32 + c];
  wdv[k * 4 + h] = s;
}

// a_d[n,h] = sum_k x[n,k] * wdv[k,h]
__global__ void k_ad(const float* __restrict__ x, const float* __restrict__ wdv,
                     float* __restrict__ ad, int n) {
  int wave = threadIdx.x >> 5, lane = threadIdx.x & 31;
  int node = blockIdx.x * 8 + wave;
  if (node >= n) return;
  float p0 = 0, p1 = 0, p2 = 0, p3 = 0;
  int k0 = lane * 4;
#pragma unroll
  for (int j = 0; j < 4; ++j) {
    float xv = x[(size_t)node * 128 + k0 + j];
    const float* w = wdv + (k0 + j) * 4;
    p0 += xv * w[0]; p1 += xv * w[1]; p2 += xv * w[2]; p3 += xv * w[3];
  }
#pragma unroll
  for (int m = 1; m < 32; m <<= 1) {
    p0 += __shfl_xor(p0, m); p1 += __shfl_xor(p1, m);
    p2 += __shfl_xor(p2, m); p3 += __shfl_xor(p3, m);
  }
  if (lane == 0) {
    ad[node * 4 + 0] = p0; ad[node * 4 + 1] = p1;
    ad[node * 4 + 2] = p2; ad[node * 4 + 3] = p3;
  }
}

__device__ __forceinline__ float edge_logit(const float* as_, const float* ad,
                                            int s, int d, int h) {
  float v = as_[s * 4 + h] + ad[d * 4 + h];
  return v > 0.f ? v : 0.2f * v; // leaky_relu 0.2
}

__global__ void k_edge_max(const float* __restrict__ as_, const float* __restrict__ ad,
                           const int* __restrict__ src, const int* __restrict__ dst,
                           unsigned* __restrict__ mu, int E) {
  int i = blockIdx.x * blockDim.x + threadIdx.x;
  if (i >= E * 4) return;
  int e = i >> 2, h = i & 3;
  float v = edge_logit(as_, ad, src[e], dst[e], h);
  atomicMax(&mu[dst[e] * 4 + h], fmax_key(v));
}

__global__ void k_edge_expsum(const float* __restrict__ as_, const float* __restrict__ ad,
                              const int* __restrict__ src, const int* __restrict__ dst,
                              const unsigned* __restrict__ mu, float* __restrict__ z,
                              float* __restrict__ alpha, int E) {
  int i = blockIdx.x * blockDim.x + threadIdx.x;
  if (i >= E * 4) return;
  int e = i >> 2, h = i & 3;
  float v = edge_logit(as_, ad, src[e], dst[e], h);
  float m = fmax_unkey(mu[dst[e] * 4 + h]);
  float ex = expf(v - m);
  alpha[i] = ex;
  atomicAdd(&z[dst[e] * 4 + h], ex);
}

// one wave per edge: acc[dst] += hs[src] * alpha  (4 floats/lane, f32 atomics)
__global__ void k_edge_scatter(const float* __restrict__ hs, const float* __restrict__ alpha,
                               const float* __restrict__ z, const int* __restrict__ src,
                               const int* __restrict__ dst, float* __restrict__ acc, int E) {
  int wave = threadIdx.x >> 5, lane = threadIdx.x & 31;
  int e = blockIdx.x * 8 + wave;
  if (e >= E) return;
  int s = src[e], d = dst[e];
  int h = lane >> 3;
  float al = alpha[e * 4 + h] / (z[d * 4 + h] + 1e-16f);
  int k0 = lane * 4;
  const float* hp = hs + (size_t)s * 128 + k0;
  float* ap = acc + (size_t)d * 128 + k0;
#pragma unroll
  for (int j = 0; j < 4; ++j) atomicAdd(&ap[j], hp[j] * al);
}

__global__ void k_bias_accum(float* __restrict__ bsum, const float* __restrict__ b) {
  int i = threadIdx.x;
  if (i < 128) bsum[i] += b[i];
}

// x = elu(acc + bsum_row); writes f32 + f16 copies
__global__ void k_elu(const float* __restrict__ acc, const float* __restrict__ bsum,
                      float* __restrict__ xf, _Float16* __restrict__ xh, int cnt) {
  int i = blockIdx.x * blockDim.x + threadIdx.x;
  if (i >= cnt) return;
  float v = acc[i] + bsum[i & 127];
  v = v > 0.f ? v : (expf(v) - 1.f);
  xf[i] = v; xh[i] = (_Float16)v;
}

// out[n] = h2[n,:] @ W3 + b3   (K=64)
__global__ void k_head(const float* __restrict__ h2, const float* __restrict__ W3,
                       const float* __restrict__ b3, float* __restrict__ out, int n) {
  int i = blockIdx.x * blockDim.x + threadIdx.x;
  if (i >= n) return;
  float s = b3[0];
  for (int k = 0; k < 64; ++k) s += h2[(size_t)i * 64 + k] * W3[k];
  out[i] = s;
}

// ---------- host orchestration ----------
extern "C" void kernel_launch(void* const* d_in, const int* in_sizes, int n_in,
                              void* d_out, int out_size, void* d_ws, size_t ws_size,
                              hipStream_t stream) {
  (void)n_in; (void)out_size; (void)ws_size;
  // node types: 0 material, 1 solution, 2 ree, 3 chemistry, 4 operation, 5 solution_num, 6 experiment
  static const int  NNT[7]  = {512, 128, 17, 100000, 100000, 100000, 100000};
  static const long ROFF[7] = {0, 512, 640, 657, 100657, 200657, 300657};
  const long TOT = 400657;
  static const int ESRC[22] = {0,1,2,3,4,5, 6,6,6,6,6,6, 0,1,5,4,0, 3,3,4,3,2};
  static const int EDST[22] = {6,6,6,6,6,6, 0,1,2,3,4,5, 3,3,4,3,2, 0,1,5,4,0};
  static const int FEATN[4] = {3,5,1,1};

  // workspace carve-up
  char* p = (char*)d_ws;
  auto take = [&](size_t bytes) -> char* {
    char* r = p; p += (bytes + 255) & ~(size_t)255; return r;
  };
  float*    xf    = (float*)take((size_t)TOT * 128 * 4);
  float*    acc   = (float*)take((size_t)TOT * 128 * 4);
  _Float16* xh    = (_Float16*)take((size_t)TOT * 128 * 2);
  float*    hs    = (float*)take((size_t)100000 * 128 * 4);   // also reused as h2 f32 (100000x64)
  _Float16* h1h   = (_Float16*)take((size_t)100000 * 128 * 2); // enc hidden / reg hidden f16
  float*    as_   = (float*)take((size_t)100000 * 4 * 4);
  float*    ad    = (float*)take((size_t)100000 * 4 * 4);
  unsigned* mu    = (unsigned*)take((size_t)100000 * 4 * 4);
  float*    z     = (float*)take((size_t)100000 * 4 * 4);
  float*    alpha = (float*)take((size_t)100000 * 4 * 4);
  float*    wdv   = (float*)take(512 * 4);
  float*    bsum  = (float*)take(7 * 128 * 4);
  _Float16* Wh    = (_Float16*)take(128 * 128 * 2);

  // ---- stage A: embeddings + encoders -> xf / xh ----
  for (int t = 0; t < 3; ++t) {
    int n = NNT[t];
    k_gather_emb<<<(n * 128 + 255) / 256, 256, 0, stream>>>(
        (const int*)d_in[t], (const float*)d_in[29 + t],
        xf + ROFF[t] * 128, xh + ROFF[t] * 128, n);
  }
  for (int j = 0; j < 4; ++j) {
    int nt = 3 + j, f = FEATN[j], n = NNT[nt];
    const float* W1 = (const float*)d_in[32 + 4 * j];
    const float* b1 = (const float*)d_in[33 + 4 * j];
    const float* W2 = (const float*)d_in[34 + 4 * j];
    const float* b2 = (const float*)d_in[35 + 4 * j];
    k_enc1<<<(n * 128 + 255) / 256, 256, 0, stream>>>(
        (const float*)d_in[3 + j], W1, b1, h1h, n, f);
    k_cvt_h<<<64, 256, 0, stream>>>(W2, Wh, 16384);
    k_gemm<1><<<(n + 127) / 128, 256, 0, stream>>>(
        h1h, Wh, b2, xf + ROFF[nt] * 128, xh + ROFF[nt] * 128, n, 128);
  }

  // ---- stage B: two hetero conv layers ----
  for (int layer = 0; layer < 2; ++layer) {
    int base = (layer == 0) ? 48 : 158;
    hipMemsetAsync(acc, 0, (size_t)TOT * 128 * 4, stream);
    hipMemsetAsync(bsum, 0, 7 * 128 * 4, stream);
    for (int e = 0; e < 22; ++e) {
      int s = ESRC[e], d = EDST[e];
      int Ns = NNT[s], Nd = NNT[d];
      int Ee = in_sizes[7 + e] / 2;
      const int* srcI = (const int*)d_in[7 + e];
      const int* dstI = srcI + Ee;
      const float* Ws   = (const float*)d_in[base + 5 * e + 0];
      const float* Wd   = (const float*)d_in[base + 5 * e + 1];
      const float* atts = (const float*)d_in[base + 5 * e + 2];
      const float* attd = (const float*)d_in[base + 5 * e + 3];
      const float* bv   = (const float*)d_in[base + 5 * e + 4];

      k_cvt_h<<<64, 256, 0, stream>>>(Ws, Wh, 16384);
      k_gemm<0><<<(Ns + 127) / 128, 256, 0, stream>>>(
          xh + ROFF[s] * 128, Wh, nullptr, hs, nullptr, Ns, 128);
      k_as<<<(Ns + 7) / 8, 256, 0, stream>>>(hs, atts, as_, Ns);
      k_wdv<<<1, 512, 0, stream>>>(Wd, attd, wdv);
      k_ad<<<(Nd + 7) / 8, 256, 0, stream>>>(xf + ROFF[d] * 128, wdv, ad, Nd);
      hipMemsetAsync(mu, 0, (size_t)Nd * 16, stream);
      hipMemsetAsync(z, 0, (size_t)Nd * 16, stream);
      k_edge_max<<<((size_t)Ee * 4 + 255) / 256, 256, 0, stream>>>(
          as_, ad, srcI, dstI, mu, Ee);
      k_edge_expsum<<<((size_t)Ee * 4 + 255) / 256, 256, 0, stream>>>(
          as_, ad, srcI, dstI, mu, z, alpha, Ee);
      k_edge_scatter<<<(Ee + 7) / 8, 256, 0, stream>>>(
          hs, alpha, z, srcI, dstI, acc + ROFF[d] * 128, Ee);
      k_bias_accum<<<1, 128, 0, stream>>>(bsum + d * 128, bv);
    }
    for (int nt = 0; nt < 7; ++nt) {
      int cnt = NNT[nt] * 128;
      k_elu<<<(cnt + 255) / 256, 256, 0, stream>>>(
          acc + ROFF[nt] * 128, bsum + nt * 128,
          xf + ROFF[nt] * 128, xh + ROFF[nt] * 128, cnt);
    }
  }

  // ---- stage C: regression head on experiment nodes ----
  const float* rW1 = (const float*)d_in[268];
  const float* rb1 = (const float*)d_in[269];
  const float* rW2 = (const float*)d_in[270];
  const float* rb2 = (const float*)d_in[271];
  const float* rW3 = (const float*)d_in[272];
  const float* rb3 = (const float*)d_in[273];
  int NE = NNT[6];
  k_cvt_h<<<64, 256, 0, stream>>>(rW1, Wh, 16384);
  k_gemm<2><<<(NE + 127) / 128, 256, 0, stream>>>(
      xh + ROFF[6] * 128, Wh, rb1, nullptr, h1h, NE, 128);
  k_cvt_h<<<32, 256, 0, stream>>>(rW2, Wh, 8192);
  k_gemm<2><<<(NE + 127) / 128, 256, 0, stream>>>(
      h1h, Wh, rb2, hs, nullptr, NE, 64);
  k_head<<<(NE + 255) / 256, 256, 0, stream>>>(hs, rW3, rb3, (float*)d_out, NE);
}
